// MambaBlock_53781580480966
// MI455X (gfx1250) — matrix operand order repaired and checked by reading, not compiled, not run on hardware
//
#include <hip/hip_runtime.h>

// MambaBlock fused: GEMM1(in_proj)+bias+silu*silu gate -> GEMM2(out_proj)+bias.
// f16 WMMA (v_wmma_f32_16x16x32_f16) with f32 accumulation; compute-bound on MI455X.
// LDS staging via gfx1250 async copies (GLOBAL_LOAD_ASYNC_TO_LDS_B128 / ASYNCcnt)
// when the toolchain declares the builtins; otherwise reg-staged fallback.

typedef __attribute__((ext_vector_type(16))) _Float16 v16h;
typedef __attribute__((ext_vector_type(8)))  _Float16 v8h;
typedef __attribute__((ext_vector_type(4)))  _Float16 v4h;
typedef __attribute__((ext_vector_type(8)))  float    v8f;
typedef __attribute__((ext_vector_type(4)))  int      v4i;

#define DM   1024          // d_model
#define BLN  32768         // B*L tokens
#define LDT  48            // LDS row stride in halfs (96B, keeps b128 chunks 16B aligned)

#if defined(__has_builtin)
#if __has_builtin(__builtin_amdgcn_global_load_async_to_lds_b128)
#define HAVE_ASYNC_LDS 1
#endif
#endif
#ifndef HAVE_ASYNC_LDS
#define HAVE_ASYNC_LDS 0
#endif

#if HAVE_ASYNC_LDS
#if __has_builtin(__builtin_amdgcn_s_wait_asynccnt)
#define WAIT_ASYNC0() __builtin_amdgcn_s_wait_asynccnt(0)
#else
#define WAIT_ASYNC0() asm volatile("s_wait_asynccnt 0x0" ::: "memory")
#endif
static __device__ __forceinline__ void async_ld16(const _Float16* g, _Float16* l) {
  __builtin_amdgcn_global_load_async_to_lds_b128(
      (__attribute__((address_space(1))) v4i*)(g),
      (__attribute__((address_space(3))) v4i*)(l), 0, 0);
}
#endif

static __device__ __forceinline__ v16h cat16(v8h lo, v8h hi) {
  return __builtin_shufflevector(lo, hi, 0,1,2,3,4,5,6,7,8,9,10,11,12,13,14,15);
}
static __device__ __forceinline__ float silu_f(float v) {
  return v / (1.0f + __expf(-v));
}

// ---------------- fp32 -> f16 conversion (vectorized) ----------------
__global__ __launch_bounds__(256) void cvt_f32_f16(const float* __restrict__ in,
                                                   _Float16* __restrict__ out, int n4) {
  int i = blockIdx.x * 256 + threadIdx.x;
  if (i < n4) {
    const float4 v = ((const float4*)in)[i];
    v4h o = { (_Float16)v.x, (_Float16)v.y, (_Float16)v.z, (_Float16)v.w };
    ((v4h*)out)[i] = o;
  }
}

// ---------------- GEMM1: x @ in_w^T + in_b, fused silu(xv)*silu(res) ----------------
// Block tile: M=128 rows x 64 gated columns. B-tile = 128 rows of in_w:
//   LDS B rows [0,64)  -> in_w rows d0+r        (x_val columns)
//   LDS B rows [64,128)-> in_w rows 1024+d0+r   (res  columns)
// 8 waves = 4 (M) x 2 (N). Wave: 2 M-tiles x 2 N-tiles x {x_val,res} = 8 accums.
__global__ __launch_bounds__(256) void gemm1_silu_gate(
    const _Float16* __restrict__ xh,    // [BLN, DM] f16
    const _Float16* __restrict__ iwh,   // [2*DM, DM] f16
    const float*    __restrict__ inb,   // [2*DM] f32
    _Float16*       __restrict__ gated) // [BLN, DM] f16
{
  __shared__ _Float16 As[2 * 128 * LDT];
  __shared__ _Float16 Bs[2 * 128 * LDT];
  const int lbuf = 128 * LDT;

  const int m0 = blockIdx.x * 128;
  const int d0 = blockIdx.y * 64;
  const int tid  = threadIdx.x;
  const int lane = tid & 31;
  const int wv   = tid >> 5;
  const int hw   = lane >> 4;   // half-wave: 0/1
  const int lm   = lane & 15;
  const int mg   = wv >> 1;     // 0..3
  const int ng   = wv & 1;      // 0..1

  // staging: each thread owns two 16B chunks per tile: rows r0 and r0+64, chunk col c0
  const int r0 = tid >> 2;      // 0..63
  const int c0 = tid & 3;       // 0..3

  const _Float16* ap0 = xh  + (size_t)(m0 + r0) * DM + c0 * 8;
  const _Float16* ap1 = ap0 + (size_t)64 * DM;
  const _Float16* bp0 = iwh + (size_t)(d0 + r0) * DM + c0 * 8;
  const _Float16* bp1 = iwh + (size_t)(DM + d0 + r0) * DM + c0 * 8;

  _Float16* asl0 = &As[ r0       * LDT + c0 * 8];
  _Float16* asl1 = &As[(r0 + 64) * LDT + c0 * 8];
  _Float16* bsl0 = &Bs[ r0       * LDT + c0 * 8];
  _Float16* bsl1 = &Bs[(r0 + 64) * LDT + c0 * 8];

  // prologue: stage K-tile 0 into buffer 0
#if HAVE_ASYNC_LDS
  async_ld16(ap0, asl0);
  async_ld16(ap1, asl1);
  async_ld16(bp0, bsl0);
  async_ld16(bp1, bsl1);
  WAIT_ASYNC0();
#else
  *(v8h*)asl0 = *(const v8h*)ap0;
  *(v8h*)asl1 = *(const v8h*)ap1;
  *(v8h*)bsl0 = *(const v8h*)bp0;
  *(v8h*)bsl1 = *(const v8h*)bp1;
#endif
  __syncthreads();

  v8f acc[2][2][2] = {};   // [mi][ni][p: 0=x_val, 1=res]

  for (int kt = 0; kt < DM / 32; ++kt) {
    const int  cur = (kt & 1) * lbuf;
    const bool pre = (kt + 1) < (DM / 32);
#if HAVE_ASYNC_LDS
    // Issue next tile's async copies immediately: the target buffer was last
    // read before the barrier that ended iteration kt-1, so it is free now.
    if (pre) {
      const int off = (kt + 1) * 32;
      const int nb  = ((kt + 1) & 1) * lbuf;
      async_ld16(ap0 + off, asl0 + nb);
      async_ld16(ap1 + off, asl1 + nb);
      async_ld16(bp0 + off, bsl0 + nb);
      async_ld16(bp1 + off, bsl1 + nb);
    }
#else
    v8h pa0, pa1, pb0, pb1;
    if (pre) {
      const int off = (kt + 1) * 32;
      pa0 = *(const v8h*)(ap0 + off);
      pa1 = *(const v8h*)(ap1 + off);
      pb0 = *(const v8h*)(bp0 + off);
      pb1 = *(const v8h*)(bp1 + off);
    }
#endif

    v16h afr[2], bfr[2][2];
#pragma unroll
    for (int mi = 0; mi < 2; ++mi) {
      const int m = mg * 32 + mi * 16 + lm;
      afr[mi] = cat16(*(const v8h*)&As[cur + m * LDT + hw * 8],
                      *(const v8h*)&As[cur + m * LDT + 16 + hw * 8]);
    }
#pragma unroll
    for (int p = 0; p < 2; ++p)
#pragma unroll
      for (int ni = 0; ni < 2; ++ni) {
        const int n = p * 64 + ng * 32 + ni * 16 + lm;
        bfr[p][ni] = cat16(*(const v8h*)&Bs[cur + n * LDT + hw * 8],
                           *(const v8h*)&Bs[cur + n * LDT + 16 + hw * 8]);
      }
#pragma unroll
    for (int mi = 0; mi < 2; ++mi)
#pragma unroll
      for (int ni = 0; ni < 2; ++ni)
#pragma unroll
        for (int p = 0; p < 2; ++p)
          acc[mi][ni][p] = __builtin_amdgcn_wmma_f32_16x16x32_f16(
              false, afr[mi], false, bfr[p][ni], (short)0, acc[mi][ni][p],
              false, false);

    if (pre) {
#if HAVE_ASYNC_LDS
      WAIT_ASYNC0();
      __syncthreads();
#else
      const int nb = ((kt + 1) & 1) * lbuf;
      __syncthreads();
      *(v8h*)(asl0 + nb) = pa0;
      *(v8h*)(asl1 + nb) = pa1;
      *(v8h*)(bsl0 + nb) = pb0;
      *(v8h*)(bsl1 + nb) = pb1;
      __syncthreads();
#endif
    }
  }

  // epilogue: bias + silu(x_val)*silu(res) -> f16 gated
#pragma unroll
  for (int mi = 0; mi < 2; ++mi)
#pragma unroll
    for (int ni = 0; ni < 2; ++ni) {
      const int dcol = d0 + ng * 32 + ni * 16 + lm;
      const float bx = inb[dcol];
      const float br = inb[DM + dcol];
      const int mbase = m0 + mg * 32 + mi * 16 + hw * 8;
      const v8f ax = acc[mi][ni][0];
      const v8f ar = acc[mi][ni][1];
#pragma unroll
      for (int v = 0; v < 8; ++v) {
        const float g = silu_f(ax[v] + bx) * silu_f(ar[v] + br);
        gated[(size_t)(mbase + v) * DM + dcol] = (_Float16)g;
      }
    }
}

// ---------------- GEMM2: gated @ out_w^T + out_b -> fp32 out ----------------
// Block tile 128x128; 8 waves = 4(M) x 2(N); wave = 2 M-tiles x 4 N-tiles.
__global__ __launch_bounds__(256) void gemm2_bias(
    const _Float16* __restrict__ gh,    // [BLN, DM] f16
    const _Float16* __restrict__ owh,   // [DM, DM] f16
    const float*    __restrict__ ob,    // [DM] f32
    float*          __restrict__ out)   // [BLN, DM] f32
{
  __shared__ _Float16 As[2 * 128 * LDT];
  __shared__ _Float16 Bs[2 * 128 * LDT];
  const int lbuf = 128 * LDT;

  const int m0 = blockIdx.x * 128;
  const int n0 = blockIdx.y * 128;
  const int tid  = threadIdx.x;
  const int lane = tid & 31;
  const int wv   = tid >> 5;
  const int hw   = lane >> 4;
  const int lm   = lane & 15;
  const int mg   = wv >> 1;     // 0..3
  const int ng   = wv & 1;      // 0..1

  const int r0 = tid >> 2;
  const int c0 = tid & 3;

  const _Float16* ap0 = gh  + (size_t)(m0 + r0) * DM + c0 * 8;
  const _Float16* ap1 = ap0 + (size_t)64 * DM;
  const _Float16* bp0 = owh + (size_t)(n0 + r0) * DM + c0 * 8;
  const _Float16* bp1 = bp0 + (size_t)64 * DM;

  _Float16* asl0 = &As[ r0       * LDT + c0 * 8];
  _Float16* asl1 = &As[(r0 + 64) * LDT + c0 * 8];
  _Float16* bsl0 = &Bs[ r0       * LDT + c0 * 8];
  _Float16* bsl1 = &Bs[(r0 + 64) * LDT + c0 * 8];

#if HAVE_ASYNC_LDS
  async_ld16(ap0, asl0);
  async_ld16(ap1, asl1);
  async_ld16(bp0, bsl0);
  async_ld16(bp1, bsl1);
  WAIT_ASYNC0();
#else
  *(v8h*)asl0 = *(const v8h*)ap0;
  *(v8h*)asl1 = *(const v8h*)ap1;
  *(v8h*)bsl0 = *(const v8h*)bp0;
  *(v8h*)bsl1 = *(const v8h*)bp1;
#endif
  __syncthreads();

  v8f acc[2][4] = {};

  for (int kt = 0; kt < DM / 32; ++kt) {
    const int  cur = (kt & 1) * lbuf;
    const bool pre = (kt + 1) < (DM / 32);
#if HAVE_ASYNC_LDS
    if (pre) {
      const int off = (kt + 1) * 32;
      const int nb  = ((kt + 1) & 1) * lbuf;
      async_ld16(ap0 + off, asl0 + nb);
      async_ld16(ap1 + off, asl1 + nb);
      async_ld16(bp0 + off, bsl0 + nb);
      async_ld16(bp1 + off, bsl1 + nb);
    }
#else
    v8h pa0, pa1, pb0, pb1;
    if (pre) {
      const int off = (kt + 1) * 32;
      pa0 = *(const v8h*)(ap0 + off);
      pa1 = *(const v8h*)(ap1 + off);
      pb0 = *(const v8h*)(bp0 + off);
      pb1 = *(const v8h*)(bp1 + off);
    }
#endif

    v16h afr[2], bfr[4];
#pragma unroll
    for (int mi = 0; mi < 2; ++mi) {
      const int m = mg * 32 + mi * 16 + lm;
      afr[mi] = cat16(*(const v8h*)&As[cur + m * LDT + hw * 8],
                      *(const v8h*)&As[cur + m * LDT + 16 + hw * 8]);
    }
#pragma unroll
    for (int ni = 0; ni < 4; ++ni) {
      const int n = ng * 64 + ni * 16 + lm;
      bfr[ni] = cat16(*(const v8h*)&Bs[cur + n * LDT + hw * 8],
                      *(const v8h*)&Bs[cur + n * LDT + 16 + hw * 8]);
    }
#pragma unroll
    for (int mi = 0; mi < 2; ++mi)
#pragma unroll
      for (int ni = 0; ni < 4; ++ni)
        acc[mi][ni] = __builtin_amdgcn_wmma_f32_16x16x32_f16(
            false, afr[mi], false, bfr[ni], (short)0, acc[mi][ni],
            false, false);

    if (pre) {
#if HAVE_ASYNC_LDS
      WAIT_ASYNC0();
      __syncthreads();
#else
      const int nb = ((kt + 1) & 1) * lbuf;
      __syncthreads();
      *(v8h*)(asl0 + nb) = pa0;
      *(v8h*)(asl1 + nb) = pa1;
      *(v8h*)(bsl0 + nb) = pb0;
      *(v8h*)(bsl1 + nb) = pb1;
      __syncthreads();
#endif
    }
  }

#pragma unroll
  for (int mi = 0; mi < 2; ++mi)
#pragma unroll
    for (int ni = 0; ni < 4; ++ni) {
      const int ncol = n0 + ng * 64 + ni * 16 + lm;
      const float bb = ob[ncol];
      const int mbase = m0 + mg * 32 + mi * 16 + hw * 8;
      const v8f a = acc[mi][ni];
#pragma unroll
      for (int v = 0; v < 8; ++v)
        out[(size_t)(mbase + v) * DM + ncol] = a[v] + bb;
    }
}

// ---------------- launch ----------------
extern "C" void kernel_launch(void* const* d_in, const int* in_sizes, int n_in,
                              void* d_out, int out_size, void* d_ws, size_t ws_size,
                              hipStream_t stream) {
  const float* x    = (const float*)d_in[0];   // [4,8192,1024]
  const float* in_w = (const float*)d_in[1];   // [2048,1024]
  const float* in_b = (const float*)d_in[2];   // [2048]
  const float* outw = (const float*)d_in[3];   // [1024,1024]
  const float* outb = (const float*)d_in[4];   // [1024]
  // d_in[5..8] (dt_w, dt_b, A, Dp) are dead compute in the reference.
  float* out = (float*)d_out;

  char* ws = (char*)d_ws;
  _Float16* xh    = (_Float16*)(ws);                          // 67,108,864 B
  _Float16* gated = (_Float16*)(ws + (size_t)67108864);       // 67,108,864 B
  _Float16* iwh   = (_Float16*)(ws + (size_t)134217728);      //  4,194,304 B
  _Float16* owh   = (_Float16*)(ws + (size_t)138412032);      //  2,097,152 B
  // total ws usage: 140,509,184 B

  const int nx  = BLN * DM;       // 33,554,432
  const int niw = 2 * DM * DM;    //  2,097,152
  const int now = DM * DM;        //  1,048,576

  cvt_f32_f16<<<(nx  / 4 + 255) / 256, 256, 0, stream>>>(x,    xh,  nx  / 4);
  cvt_f32_f16<<<(niw / 4 + 255) / 256, 256, 0, stream>>>(in_w, iwh, niw / 4);
  cvt_f32_f16<<<(now / 4 + 255) / 256, 256, 0, stream>>>(outw, owh, now / 4);

  gemm1_silu_gate<<<dim3(BLN / 128, DM / 64), 256, 0, stream>>>(xh, iwh, in_b, gated);
  gemm2_bias     <<<dim3(BLN / 128, DM / 128), 256, 0, stream>>>(gated, owh, outb, out);
}
